// HomeostaticEqProp_46084999086213
// MI455X (gfx1250) — compile-verified, matrix operand order checked by reading, not run on hardware
//
#include <hip/hip_runtime.h>
#include <math.h>

// Problem constants (fixed by the reference).
#define B_SZ   8192
#define IN_D   256
#define H_D    256
#define OUT_D  128
#define L_N    5
#define STEPS  30
#define ALPHA  0.5f

// LDS row pitch for the staged 64x256 A tile. 264 bf16 = 528 B = 132 dwords,
// i.e. a 4-bank rotation per row across the 64 LDS banks -> the 16-lane
// fragment read (one 16B chunk per row) is bank-conflict free. 528 % 16 == 0
// keeps every chunk 16B aligned.
#define LDS_PITCH 264

typedef __attribute__((ext_vector_type(16))) __bf16 v16bf;
typedef __attribute__((ext_vector_type(8)))  __bf16 v8bf;
typedef __attribute__((ext_vector_type(8)))  float  v8f;

// ---------------------------------------------------------------------------
// Weight packer: W[N][K] (row major, fp32, optional per-layer scale) ->
// fragment-major bf16 so a wave reads its B fragment as 32 contiguous
// bytes per lane. Packed index = (((nt*Ktiles + kt)*32 + lane)*16 + e).
// B layout mirrors A (B[k][n], n = lane&15, same k mapping), per the CDNA5
// 16-bit WMMA operand layout.
// ---------------------------------------------------------------------------
__global__ void pack_w_kernel(const float* __restrict__ W,
                              const float* __restrict__ scales,
                              int L, int N, int K, __bf16* __restrict__ out) {
  int idx = blockIdx.x * blockDim.x + threadIdx.x;
  int total = L * N * K;
  if (idx >= total) return;
  int per_layer = N * K;
  int l = idx / per_layer;
  int r = idx - l * per_layer;
  int e    = r & 15;
  int lane = (r >> 4) & 31;
  int t    = r >> 9;            // tile index = nt*Ktiles + kt
  int Ktiles = K >> 5;
  int kt = t % Ktiles;
  int nt = t / Ktiles;
  int k = kt * 32 + ((e < 8) ? 0 : 16) + ((lane >= 16) ? 8 : 0) + (e & 7);
  int n = nt * 16 + (lane & 15);
  float s = scales ? scales[l] : 1.0f;
  out[idx] = (__bf16)(W[(size_t)l * per_layer + (size_t)n * K + k] * s);
}

__global__ void convert_f32_to_bf16_kernel(const float* __restrict__ src,
                                           __bf16* __restrict__ dst, int n) {
  int i = blockIdx.x * blockDim.x + threadIdx.x;
  if (i < n) dst[i] = (__bf16)src[i];
}

__global__ void fill_zero_u32_kernel(unsigned int* __restrict__ p, int n) {
  int i = blockIdx.x * blockDim.x + threadIdx.x;
  int stride = gridDim.x * blockDim.x;
  for (; i < n; i += stride) p[i] = 0u;
}

// ---------------------------------------------------------------------------
// Shared GEMM core for a 64-row x (NT waves' columns) tile, K = KTILES*32.
// 1) Cooperatively stage the 64xK A tile (bf16, row major) into padded LDS:
//    256 threads x 64 contiguous elements each (global_load_b128+ds_store).
// 2) One barrier.
// 3) Fully-unrolled kt loop: A fragments from LDS (ds_load_b128 x2,
//    conflict-free thanks to the 264-element pitch), B fragments streamed from
//    the L2-resident fragment-major packed weights (global_load_b128 x2,
//    contiguous 32B/lane), 4*NT v_wmma_f32_16x16x32_bf16 per kt per wave.
//    Full unroll lets the scheduler hoist the next k-step's loads above the
//    current WMMA group (software pipelining) instead of stalling on
//    s_wait_loadcnt 0 every iteration.
// A-fragment layout per CDNA5 ISA (16-bit A 16x32):
//   lane 0-15 : M=lane,    K={0..7,16..23}; lane 16-31: M=lane-16, K={8..15,24..31}
// ---------------------------------------------------------------------------
template<int NT, int KTILES>
__device__ __forceinline__ void gemm_core(const __bf16* __restrict__ A, int lda,
                                          const __bf16* __restrict__ Wp,
                                          int m0, int nt0,
                                          __bf16* lds, int tid,
                                          v8f (&acc)[4][NT]) {
  // ---- stage A tile: row = tid/4, 64-element chunk = tid%4 (K must be 256)
  {
    const int row  = tid >> 2;
    const int col0 = (tid & 3) * 64;
    const __bf16* g = A + (size_t)(m0 + row) * lda + col0;
    __bf16* s = lds + row * LDS_PITCH + col0;
#pragma unroll
    for (int c = 0; c < 8; ++c)
      *(v8bf*)(s + c * 8) = *(const v8bf*)(g + c * 8);
  }
  __syncthreads();

  const int lane = tid & 31;
#pragma unroll
  for (int kt = 0; kt < KTILES; ++kt) {
    const int kb = kt * 32;
    v16bf a[4];
#pragma unroll
    for (int mt = 0; mt < 4; ++mt) {
      const __bf16* p = lds + (mt * 16 + (lane & 15)) * LDS_PITCH
                            + kb + ((lane >> 4) << 3);
      v8bf lo = *(const v8bf*)(p);        // K + 0..7
      v8bf hi = *(const v8bf*)(p + 16);   // K + 16..23
#pragma unroll
      for (int i = 0; i < 8; ++i) { a[mt][i] = lo[i]; a[mt][8 + i] = hi[i]; }
    }
#pragma unroll
    for (int ntl = 0; ntl < NT; ++ntl) {
      v16bf b = *(const v16bf*)(Wp + (((size_t)(nt0 + ntl) * KTILES + kt) * 32 + lane) * 16);
#pragma unroll
      for (int mt = 0; mt < 4; ++mt)
        acc[mt][ntl] = __builtin_amdgcn_wmma_f32_16x16x32_bf16(
            false, a[mt], false, b, (short)0, acc[mt][ntl], false, false);
    }
  }
}

// ---------------------------------------------------------------------------
// Embedding GEMM: x_emb = x_bf16 @ W_in^T + b_in  -> bf16 [B, H]
// Block = 256 threads (8 waves). Wave w: cols [w*32, w*32+32), rows m0..m0+63.
// ---------------------------------------------------------------------------
__global__ __launch_bounds__(256)
void emb_gemm_kernel(const __bf16* __restrict__ A,      // [B, IN] bf16
                     const __bf16* __restrict__ Wp,     // packed W_in bf16
                     const float*  __restrict__ bias,   // [H]
                     __bf16* __restrict__ out) {        // [B, H] bf16
  __shared__ __bf16 smem[64 * LDS_PITCH];
  const int m0   = blockIdx.x * 64;
  const int tid  = threadIdx.x;
  const int wave = tid >> 5;
  const int lane = tid & 31;

  v8f acc[4][2] = {};
  gemm_core<2, IN_D / 32>(A, IN_D, Wp, m0, wave * 2, smem, tid, acc);

#pragma unroll
  for (int ntl = 0; ntl < 2; ++ntl) {
    const int n = wave * 32 + ntl * 16 + (lane & 15);
    const float bn = bias[n];
#pragma unroll
    for (int mt = 0; mt < 4; ++mt) {
      const int mb = m0 + mt * 16 + ((lane >> 4) << 3);
#pragma unroll
      for (int r = 0; r < 8; ++r)
        out[(size_t)(mb + r) * H_D + n] = (__bf16)(acc[mt][ntl][r] + bn);
    }
  }
}

// ---------------------------------------------------------------------------
// One Jacobi step for all layers: grid.y = layer.
//   pre = (l==0) ? x_emb : h_prev[l-1]
//   h_next[l] = (1-a)*h_prev[l] + a*tanh(pre @ Wsc[l]^T + bs[l])
// ---------------------------------------------------------------------------
__global__ __launch_bounds__(256)
void step_kernel(const __bf16* __restrict__ x_emb,   // [B, H]
                 const __bf16* __restrict__ h_prev,  // [L, B, H]
                 __bf16*       __restrict__ h_next,  // [L, B, H]
                 const __bf16* __restrict__ Wp_ws,   // packed [L, H*H]
                 const float*  __restrict__ bs) {    // [L, H]
  __shared__ __bf16 smem[64 * LDS_PITCH];
  const int l    = blockIdx.y;
  const int m0   = blockIdx.x * 64;
  const int tid  = threadIdx.x;
  const int wave = tid >> 5;
  const int lane = tid & 31;

  const __bf16* A  = (l == 0) ? x_emb : h_prev + (size_t)(l - 1) * B_SZ * H_D;
  const __bf16* Wp = Wp_ws + (size_t)l * H_D * H_D;
  const __bf16* Hp = h_prev + (size_t)l * B_SZ * H_D;
  __bf16*       Hn = h_next + (size_t)l * B_SZ * H_D;
  const float*  bias = bs + l * H_D;

  v8f acc[4][2] = {};
  gemm_core<2, H_D / 32>(A, H_D, Wp, m0, wave * 2, smem, tid, acc);

#pragma unroll
  for (int ntl = 0; ntl < 2; ++ntl) {
    const int n = wave * 32 + ntl * 16 + (lane & 15);
    const float bn = bias[n];
#pragma unroll
    for (int mt = 0; mt < 4; ++mt) {
      const int mb = m0 + mt * 16 + ((lane >> 4) << 3);
#pragma unroll
      for (int r = 0; r < 8; ++r) {
        const size_t off = (size_t)(mb + r) * H_D + n;
        const float t  = tanhf(acc[mt][ntl][r] + bn);
        const float hp = (float)Hp[off];
        Hn[off] = (__bf16)((1.0f - ALPHA) * hp + ALPHA * t);
      }
    }
  }
}

// ---------------------------------------------------------------------------
// Head GEMM: out = h_final @ W_head^T + b_head -> fp32 [B, OUT]
// 8 waves x 1 n-tile each covers OUT=128 columns.
// ---------------------------------------------------------------------------
__global__ __launch_bounds__(256)
void head_gemm_kernel(const __bf16* __restrict__ A,      // [B, H] bf16 (h[L-1])
                      const __bf16* __restrict__ Wp,     // packed W_head
                      const float*  __restrict__ bias,   // [OUT]
                      float* __restrict__ out) {         // [B, OUT] fp32
  __shared__ __bf16 smem[64 * LDS_PITCH];
  const int m0   = blockIdx.x * 64;
  const int tid  = threadIdx.x;
  const int wave = tid >> 5;
  const int lane = tid & 31;

  v8f acc[4][1] = {};
  gemm_core<1, H_D / 32>(A, H_D, Wp, m0, wave, smem, tid, acc);

  const int n = wave * 16 + (lane & 15);
  const float bn = bias[n];
#pragma unroll
  for (int mt = 0; mt < 4; ++mt) {
    const int mb = m0 + mt * 16 + ((lane >> 4) << 3);
#pragma unroll
    for (int r = 0; r < 8; ++r)
      out[(size_t)(mb + r) * OUT_D + n] = acc[mt][0][r] + bn;
  }
}

// ---------------------------------------------------------------------------
// Host launcher. Inputs: x, W_in, b_in, Ws, bs, scales, W_head, b_head, steps
// ---------------------------------------------------------------------------
extern "C" void kernel_launch(void* const* d_in, const int* in_sizes, int n_in,
                              void* d_out, int out_size, void* d_ws, size_t ws_size,
                              hipStream_t stream) {
  const float* x       = (const float*)d_in[0];
  const float* W_in    = (const float*)d_in[1];
  const float* b_in    = (const float*)d_in[2];
  const float* Ws      = (const float*)d_in[3];
  const float* bs      = (const float*)d_in[4];
  const float* scales  = (const float*)d_in[5];
  const float* W_head  = (const float*)d_in[6];
  const float* b_head  = (const float*)d_in[7];
  (void)in_sizes; (void)n_in; (void)out_size; (void)ws_size;

  // Workspace carve-up (all offsets naturally 256B-aligned).
  char* ws = (char*)d_ws;
  __bf16* x_bf   = (__bf16*)ws;  ws += (size_t)B_SZ * IN_D * sizeof(__bf16);        // 4 MiB
  __bf16* x_emb  = (__bf16*)ws;  ws += (size_t)B_SZ * H_D  * sizeof(__bf16);        // 4 MiB
  __bf16* hA     = (__bf16*)ws;  ws += (size_t)L_N * B_SZ * H_D * sizeof(__bf16);   // 20 MiB
  __bf16* hB     = (__bf16*)ws;  ws += (size_t)L_N * B_SZ * H_D * sizeof(__bf16);   // 20 MiB
  __bf16* WpIn   = (__bf16*)ws;  ws += (size_t)H_D * IN_D * sizeof(__bf16);
  __bf16* WpWs   = (__bf16*)ws;  ws += (size_t)L_N * H_D * H_D * sizeof(__bf16);
  __bf16* WpHead = (__bf16*)ws;  ws += (size_t)OUT_D * H_D * sizeof(__bf16);

  // 1) x -> bf16
  {
    int n = B_SZ * IN_D;
    convert_f32_to_bf16_kernel<<<(n + 255) / 256, 256, 0, stream>>>(x, x_bf, n);
  }
  // 2) pack weights into fragment-major bf16
  {
    int n = H_D * IN_D;
    pack_w_kernel<<<(n + 255) / 256, 256, 0, stream>>>(W_in, nullptr, 1, H_D, IN_D, WpIn);
    n = L_N * H_D * H_D;
    pack_w_kernel<<<(n + 255) / 256, 256, 0, stream>>>(Ws, scales, L_N, H_D, H_D, WpWs);
    n = OUT_D * H_D;
    pack_w_kernel<<<(n + 255) / 256, 256, 0, stream>>>(W_head, nullptr, 1, OUT_D, H_D, WpHead);
  }
  // 3) zero initial h state (hA); hB is fully overwritten by step 1
  {
    int n_u32 = (int)((size_t)L_N * B_SZ * H_D * sizeof(__bf16) / 4);
    fill_zero_u32_kernel<<<2048, 256, 0, stream>>>((unsigned int*)hA, n_u32);
  }
  // 4) embedding GEMM
  emb_gemm_kernel<<<dim3(B_SZ / 64), 256, 0, stream>>>(x_bf, WpIn, b_in, x_emb);

  // 5) 30 Jacobi steps, ping-pong h buffers (global barrier = kernel boundary)
  __bf16* hp = hA;
  __bf16* hn = hB;
  for (int s = 0; s < STEPS; ++s) {
    step_kernel<<<dim3(B_SZ / 64, L_N), 256, 0, stream>>>(x_emb, hp, hn, WpWs, bs);
    __bf16* t = hp; hp = hn; hn = t;   // hp now holds the newest state
  }

  // 6) head GEMM from h[L-1] of the newest state
  head_gemm_kernel<<<dim3(B_SZ / 64), 256, 0, stream>>>(
      hp + (size_t)(L_N - 1) * B_SZ * H_D, WpHead, b_head, (float*)d_out);
}
